// MambaEncoder_64123861729636
// MI455X (gfx1250) — compile-verified, matrix operand order checked by reading, not run on hardware
//
#include <hip/hip_runtime.h>

// ---------------------------------------------------------------------------
// Mamba encoder for MI455X (gfx1250), wave32 + WMMA bf16.
//
// Round-1 fix: GEMMs now consume pre-converted bf16 operands so the inner
// loop is pure {2x b128 load per fragment + v_wmma_f32_16x16x32_bf16}; the
// fp32->bf16 rounding happens exactly once per element in converter /
// producer kernels instead of once per tile-use (was ~240 VALU per WMMA).
// Working set (~215 MB) is L2-resident (192MB L2 + HBM @ 23.3 TB/s), so the
// GEMMs are matrix-pipe bound; the selective scan stays latency-bound and
// register-resident with LDS broadcast of B/C.
// ---------------------------------------------------------------------------

#define D_MODEL 512
#define D_INNER 1024
#define D_STATE 16
#define D_CONV 4
#define DT_RANK 32
#define N_LAYERS 2
#define EMBED 128
#define DKEY 256
#define DVAL 256
#define BATCH 2
#define SEQ 4096
#define NTOK (BATCH * SEQ)
#define NXD (DT_RANK + 2 * D_STATE)  // 80

typedef __bf16 bf16_t;
typedef bf16_t v16bf __attribute__((ext_vector_type(16)));
typedef bf16_t v8bf __attribute__((ext_vector_type(8)));
typedef float v8f __attribute__((ext_vector_type(8)));

// fp32 -> bf16 round-to-nearest-even, pure integer.
__device__ __forceinline__ bf16_t f2bf(float f) {
  union { float f; unsigned u; } v;
  v.f = f;
  unsigned r = v.u + 0x7FFFu + ((v.u >> 16) & 1u);
  unsigned short s = (unsigned short)(r >> 16);
  return __builtin_bit_cast(bf16_t, s);
}

__device__ __forceinline__ float silu_f(float x) {
  return x / (1.0f + __expf(-x));
}

// Per-lane K-major WMMA fragment: halves 0..7 at p, halves 8..15 at p+16.
// p = row_base + k0 + hi*8 (16B aligned) -> two 16-byte loads.
__device__ __forceinline__ v16bf load_frag(const bf16_t* __restrict__ p) {
  v8bf lo = *(const v8bf*)(p);
  v8bf hh = *(const v8bf*)(p + 16);
  return __builtin_shufflevector(lo, hh, 0, 1, 2, 3, 4, 5, 6, 7,
                                 8, 9, 10, 11, 12, 13, 14, 15);
}

// ---------------------------------------------------------------------------
// GEMM: out[m,n] = sum_k A[m,k]*W[n,k] (+bias[n]) (+residual[m,n]); optional
// bf16 mirror of the output. One wave: 16M x (16*NT)N strip, A frag reused.
// K % 32 == 0, M % 16 == 0, N % (16*NT) == 0. act: 0 none, 1 softplus.
// ---------------------------------------------------------------------------
template <int NT>
__global__ __launch_bounds__(128) void gemm_bf16_wmma(
    const bf16_t* __restrict__ A, int lda,
    const bf16_t* __restrict__ W, int ldw,
    const float* __restrict__ bias,
    const float* __restrict__ residual,
    float* __restrict__ out, bf16_t* __restrict__ out_bf, int ldo,
    int M, int N, int K, int act) {
  const int lane = threadIdx.x & 31;
  const int m = lane & 15;   // row-in-tile (A), col-in-tile (B/C)
  const int hi = lane >> 4;  // K-half selector
  const int mTiles = M >> 4;
  const int nGroups = (N >> 4) / NT;
  const int waveId = blockIdx.x * 4 + (threadIdx.x >> 5);
  if (waveId >= mTiles * nGroups) return;
  const int mT = waveId % mTiles;
  const int ng = waveId / mTiles;

  v8f acc[NT];
#pragma unroll
  for (int t = 0; t < NT; ++t)
#pragma unroll
    for (int r = 0; r < 8; ++r) acc[t][r] = 0.0f;

  const bf16_t* aRow = A + (size_t)(mT * 16 + m) * lda + hi * 8;
  const bf16_t* wRow0 = W + (size_t)(ng * NT * 16 + m) * ldw + hi * 8;
  for (int k0 = 0; k0 < K; k0 += 32) {
    v16bf af = load_frag(aRow + k0);
#pragma unroll
    for (int t = 0; t < NT; ++t) {
      v16bf bf = load_frag(wRow0 + (size_t)t * 16 * ldw + k0);
      acc[t] = __builtin_amdgcn_wmma_f32_16x16x32_bf16(
          false, af, false, bf, (short)0, acc[t], false, false);
    }
  }

#pragma unroll
  for (int t = 0; t < NT; ++t) {
    const int n = (ng * NT + t) * 16 + m;
    const float bv = bias ? bias[n] : 0.0f;
#pragma unroll
    for (int r = 0; r < 8; ++r) {
      const int row = mT * 16 + r + 8 * hi;  // C layout: M = r + 8*hi
      float v = acc[t][r] + bv;
      if (residual) v += residual[(size_t)row * ldo + n];
      if (act == 1) v = (v > 20.0f) ? v : log1pf(__expf(v));
      out[(size_t)row * ldo + n] = v;
      if (out_bf) out_bf[(size_t)row * ldo + n] = f2bf(v);
    }
  }
}

// ---------------------------------------------------------------------------
// fp32 -> bf16 bulk converter, 8 elements per thread (b128 in, b128 out).
// ---------------------------------------------------------------------------
__global__ __launch_bounds__(256) void cvt_bf16_k(const float* __restrict__ src,
                                                  bf16_t* __restrict__ dst,
                                                  long n) {
  const long i = ((long)blockIdx.x * 256 + threadIdx.x) * 8;
  if (i >= n) return;
  float4 a = *(const float4*)(src + i);
  float4 b = *(const float4*)(src + i + 4);
  v8bf o;
  o[0] = f2bf(a.x); o[1] = f2bf(a.y); o[2] = f2bf(a.z); o[3] = f2bf(a.w);
  o[4] = f2bf(b.x); o[5] = f2bf(b.y); o[6] = f2bf(b.z); o[7] = f2bf(b.w);
  *(v8bf*)(dst + i) = o;
}

// ---------------------------------------------------------------------------
// LayerNorm over rows of 512, emits bf16; one block (256 threads) per row.
// ---------------------------------------------------------------------------
__global__ __launch_bounds__(256) void layernorm_k(
    const float* __restrict__ h, const float* __restrict__ w,
    const float* __restrict__ b, bf16_t* __restrict__ out) {
  const int row = blockIdx.x;
  const int tid = threadIdx.x;
  const float* hp = h + (size_t)row * D_MODEL;
  float v0 = hp[tid], v1 = hp[tid + 256];
  __shared__ float s1[256], s2[256];
  s1[tid] = v0 + v1;
  s2[tid] = v0 * v0 + v1 * v1;
  __syncthreads();
  for (int off = 128; off > 0; off >>= 1) {
    if (tid < off) { s1[tid] += s1[tid + off]; s2[tid] += s2[tid + off]; }
    __syncthreads();
  }
  const float mu = s1[0] * (1.0f / D_MODEL);
  const float var = s2[0] * (1.0f / D_MODEL) - mu * mu;
  const float rs = rsqrtf(var + 1e-5f);
  bf16_t* op = out + (size_t)row * D_MODEL;
  op[tid] = f2bf((v0 - mu) * rs * w[tid] + b[tid]);
  op[tid + 256] = f2bf((v1 - mu) * rs * w[tid + 256] + b[tid + 256]);
}

// ---------------------------------------------------------------------------
// Depthwise causal conv1d (width 4) + SiLU; emits fp32 (scan) + bf16 (GEMM).
// ---------------------------------------------------------------------------
__global__ __launch_bounds__(256) void conv_silu_k(
    const float* __restrict__ xz, const float* __restrict__ cw,
    const float* __restrict__ cb, float* __restrict__ xi,
    bf16_t* __restrict__ xibf) {
  const size_t idx = (size_t)blockIdx.x * 256 + threadIdx.x;
  const int d = (int)(idx & (D_INNER - 1));
  const size_t row = idx >> 10;  // b*SEQ + l
  const int l = (int)(row & (SEQ - 1));
  const size_t bBase = row - l;
  float acc = cb[d];
#pragma unroll
  for (int t = 0; t < D_CONV; ++t) {
    const int ls = l - (D_CONV - 1) + t;
    if (ls >= 0)
      acc += cw[d * D_CONV + t] * xz[(bBase + ls) * (2 * D_INNER) + d];
  }
  const float v = silu_f(acc);
  xi[idx] = v;
  xibf[idx] = f2bf(v);
}

// ---------------------------------------------------------------------------
// Selective scan: thread owns one (b,d); state in registers, B/C via LDS.
// y written in place into xi.
// ---------------------------------------------------------------------------
__global__ __launch_bounds__(256) void scan_k(
    const float* __restrict__ dt, const float* __restrict__ xdbl,
    float* __restrict__ xi, const float* __restrict__ A_log,
    const float* __restrict__ Dp) {
  const int bb = blockIdx.x >> 2;
  const int d = ((blockIdx.x & 3) << 8) + threadIdx.x;
  __shared__ float bc[32];  // B[0..15], C[0..15]
  float A[D_STATE], h[D_STATE];
#pragma unroll
  for (int n = 0; n < D_STATE; ++n) {
    A[n] = -__expf(A_log[d * D_STATE + n]);
    h[n] = 0.0f;
  }
  const float dpar = Dp[d];
  const float* dtp = dt + (size_t)bb * SEQ * D_INNER + d;
  float* xip = xi + (size_t)bb * SEQ * D_INNER + d;
  const float* xdp = xdbl + (size_t)bb * SEQ * NXD + DT_RANK;

  for (int l = 0; l < SEQ; ++l) {
    if (threadIdx.x < 32) bc[threadIdx.x] = xdp[(size_t)l * NXD + threadIdx.x];
    __syncthreads();
    const float dtt = dtp[(size_t)l * D_INNER];
    const float xt = xip[(size_t)l * D_INNER];
    const float dtx = dtt * xt;
    float acc = 0.0f;
#pragma unroll
    for (int n = 0; n < D_STATE; ++n) {
      const float dA = __expf(dtt * A[n]);
      h[n] = dA * h[n] + dtx * bc[n];
      acc += h[n] * bc[16 + n];
    }
    xip[(size_t)l * D_INNER] = acc + xt * dpar;
    __syncthreads();
  }
}

// ---------------------------------------------------------------------------
// ybf = bf16( y * silu(z) ), z = xz[:, D_INNER + d]
// ---------------------------------------------------------------------------
__global__ __launch_bounds__(256) void gate_k(const float* __restrict__ y,
                                              const float* __restrict__ xz,
                                              bf16_t* __restrict__ ybf) {
  const size_t idx = (size_t)blockIdx.x * 256 + threadIdx.x;
  const int d = (int)(idx & (D_INNER - 1));
  const size_t row = idx >> 10;
  const float z = xz[row * (2 * D_INNER) + D_INNER + d];
  ybf[idx] = f2bf(y[idx] * silu_f(z));
}

// ---------------------------------------------------------------------------
// Host orchestration
// ---------------------------------------------------------------------------
static inline void launch_cvt(const float* s, bf16_t* d, long n, hipStream_t st) {
  cvt_bf16_k<<<(int)((n / 8 + 255) / 256), 256, 0, st>>>(s, d, n);
}

static inline void launch_gemm4(const bf16_t* A, int lda, const bf16_t* W,
                                int ldw, const float* bias, const float* res,
                                float* o, bf16_t* obf, int ldo, int M, int N,
                                int K, int act, hipStream_t s) {
  const int waves = (M / 16) * ((N / 16) / 4);
  gemm_bf16_wmma<4><<<(waves + 3) / 4, 128, 0, s>>>(A, lda, W, ldw, bias, res,
                                                    o, obf, ldo, M, N, K, act);
}

extern "C" void kernel_launch(void* const* d_in, const int* in_sizes, int n_in,
                              void* d_out, int out_size, void* d_ws,
                              size_t ws_size, hipStream_t stream) {
  (void)in_sizes; (void)n_in; (void)out_size; (void)ws_size;
  const float* x      = (const float*)d_in[0];
  const float* ipw    = (const float*)d_in[1];
  const float* ipb    = (const float*)d_in[2];
  const float* lnw    = (const float*)d_in[3];
  const float* lnb    = (const float*)d_in[4];
  const float* inw    = (const float*)d_in[5];
  const float* cw     = (const float*)d_in[6];
  const float* cb     = (const float*)d_in[7];
  const float* xpw    = (const float*)d_in[8];
  const float* dtw    = (const float*)d_in[9];
  const float* dtbias = (const float*)d_in[10];
  const float* Alog   = (const float*)d_in[11];
  const float* Dpar   = (const float*)d_in[12];
  const float* outw   = (const float*)d_in[13];
  const float* kw     = (const float*)d_in[14];
  const float* kb     = (const float*)d_in[15];
  const float* vw     = (const float*)d_in[16];
  const float* vb     = (const float*)d_in[17];
  float* out = (float*)d_out;

  // ---- workspace carve-up ----
  // fp32 region
  float* wsf  = (float*)d_ws;
  float* h    = wsf;                               // [NTOK,512]
  float* xz   = h    + (size_t)NTOK * D_MODEL;     // [NTOK,2048]
  float* xi   = xz   + (size_t)NTOK * 2 * D_INNER; // [NTOK,1024]
  float* dtb  = xi   + (size_t)NTOK * D_INNER;     // [NTOK,1024]
  float* xdbl = dtb  + (size_t)NTOK * D_INNER;     // [NTOK,80]
  // bf16 region
  bf16_t* wsb    = (bf16_t*)(xdbl + (size_t)NTOK * NXD);
  bf16_t* xbf    = wsb;                                  // [NTOK,128]
  bf16_t* xnbf   = xbf    + (size_t)NTOK * EMBED;        // [NTOK,512]
  bf16_t* xibf   = xnbf   + (size_t)NTOK * D_MODEL;      // [NTOK,1024]
  bf16_t* ybf    = xibf   + (size_t)NTOK * D_INNER;      // [NTOK,1024]
  bf16_t* hbf    = ybf    + (size_t)NTOK * D_INNER;      // [NTOK,512]
  bf16_t* xdblbf = hbf    + (size_t)NTOK * D_MODEL;      // [NTOK,80]
  bf16_t* ipwb   = xdblbf + (size_t)NTOK * NXD;
  bf16_t* inwb   = ipwb   + (size_t)D_MODEL * EMBED;
  bf16_t* xpwb   = inwb   + (size_t)N_LAYERS * 2 * D_INNER * D_MODEL;
  bf16_t* dtwb   = xpwb   + (size_t)N_LAYERS * NXD * D_INNER;
  bf16_t* outwb  = dtwb   + (size_t)N_LAYERS * D_INNER * DT_RANK;
  bf16_t* kwb    = outwb  + (size_t)N_LAYERS * D_MODEL * D_INNER;
  bf16_t* vwb    = kwb    + (size_t)DKEY * D_MODEL;

  // ---- one-shot bf16 conversions (weights + input) ----
  launch_cvt(x, xbf, (long)NTOK * EMBED, stream);
  launch_cvt(ipw, ipwb, (long)D_MODEL * EMBED, stream);
  launch_cvt(inw, inwb, (long)N_LAYERS * 2 * D_INNER * D_MODEL, stream);
  launch_cvt(xpw, xpwb, (long)N_LAYERS * NXD * D_INNER, stream);
  launch_cvt(dtw, dtwb, (long)N_LAYERS * D_INNER * DT_RANK, stream);
  launch_cvt(outw, outwb, (long)N_LAYERS * D_MODEL * D_INNER, stream);
  launch_cvt(kw, kwb, (long)DKEY * D_MODEL, stream);
  launch_cvt(vw, vwb, (long)DVAL * D_MODEL, stream);

  // h = x @ input_proj_w^T + b
  launch_gemm4(xbf, EMBED, ipwb, EMBED, ipb, nullptr, h, nullptr, D_MODEL,
               NTOK, D_MODEL, EMBED, 0, stream);

  for (int layer = 0; layer < N_LAYERS; ++layer) {
    layernorm_k<<<NTOK, 256, 0, stream>>>(h, lnw + layer * D_MODEL,
                                          lnb + layer * D_MODEL, xnbf);
    // in_proj -> xz
    launch_gemm4(xnbf, D_MODEL, inwb + (size_t)layer * 2 * D_INNER * D_MODEL,
                 D_MODEL, nullptr, nullptr, xz, nullptr, 2 * D_INNER,
                 NTOK, 2 * D_INNER, D_MODEL, 0, stream);
    // depthwise causal conv + silu -> xi (fp32) + xibf
    conv_silu_k<<<(NTOK * D_INNER) / 256, 256, 0, stream>>>(
        xz, cw + (size_t)layer * D_INNER * D_CONV, cb + layer * D_INNER, xi,
        xibf);
    // x_proj -> xdbl (fp32 for scan B/C) + xdblbf (for dt GEMM)
    {
      const int waves = (NTOK / 16) * ((NXD / 16) / 5);
      gemm_bf16_wmma<5><<<(waves + 3) / 4, 128, 0, stream>>>(
          xibf, D_INNER, xpwb + (size_t)layer * NXD * D_INNER, D_INNER,
          nullptr, nullptr, xdbl, xdblbf, NXD, NTOK, NXD, D_INNER, 0);
    }
    // dt = softplus(xdbl[:, :32] @ dt_proj_w^T + dt_proj_b)
    launch_gemm4(xdblbf, NXD, dtwb + (size_t)layer * D_INNER * DT_RANK,
                 DT_RANK, dtbias + layer * D_INNER, nullptr, dtb, nullptr,
                 D_INNER, NTOK, D_INNER, DT_RANK, 1, stream);
    // selective scan: xi (x) -> xi (y)
    scan_k<<<BATCH * (D_INNER / 256), 256, 0, stream>>>(
        dtb, xdbl, xi, Alog + (size_t)layer * D_INNER * D_STATE,
        Dpar + layer * D_INNER);
    // gate -> ybf
    gate_k<<<(NTOK * D_INNER) / 256, 256, 0, stream>>>(xi, xz, ybf);
    // out_proj + residual -> h (fp32) + hbf (for key/val GEMMs)
    launch_gemm4(ybf, D_INNER, outwb + (size_t)layer * D_MODEL * D_INNER,
                 D_INNER, nullptr, h, h, hbf, D_MODEL,
                 NTOK, D_MODEL, D_INNER, 0, stream);
  }

  // keys then values, concatenated flat in d_out.
  launch_gemm4(hbf, D_MODEL, kwb, D_MODEL, kb, nullptr, out, nullptr, DKEY,
               NTOK, DKEY, D_MODEL, 0, stream);
  launch_gemm4(hbf, D_MODEL, vwb, D_MODEL, vb, nullptr,
               out + (size_t)NTOK * DKEY, nullptr, DVAL,
               NTOK, DVAL, D_MODEL, 0, stream);
}
